// Head_40346922779375
// MI455X (gfx1250) — matrix operand order repaired and checked by reading, not compiled, not run on hardware
//
#include <hip/hip_runtime.h>
#include <stdint.h>

typedef __attribute__((ext_vector_type(16))) __bf16        v16bf;
typedef __attribute__((ext_vector_type(4)))  __bf16        v4bf;
typedef __attribute__((ext_vector_type(8)))  float          v8f;
typedef __attribute__((ext_vector_type(16))) unsigned short v16u;
typedef __attribute__((ext_vector_type(8)))  unsigned short v8u;

#define BB 4
#define TT 4096
#define CC 1024
#define DD 64

// float -> bf16 via hardware convert (RNE)
__device__ __forceinline__ unsigned short f2bf(float f) {
  __bf16 h = (__bf16)f;
  return __builtin_bit_cast(unsigned short, h);
}

__device__ __forceinline__ v16bf mkfrag(v8u lo, v8u hi) {
  v16u t = __builtin_shufflevector(lo, hi, 0,1,2,3,4,5,6,7,8,9,10,11,12,13,14,15);
  return __builtin_bit_cast(v16bf, t);
}

// ---------------------------------------------------------------------------
// Kernel 1: W[c][d] (fp32) -> Wt[m][d][c] (bf16), m = {q,k,v}
// ---------------------------------------------------------------------------
__global__ void prep_w_kernel(const float* __restrict__ Wq,
                              const float* __restrict__ Wk,
                              const float* __restrict__ Wv,
                              unsigned short* __restrict__ wt) {
  int i = blockIdx.x * blockDim.x + threadIdx.x;   // 3*64*1024 total
  if (i >= 3 * DD * CC) return;
  int m   = i / (DD * CC);
  int rem = i % (DD * CC);
  int d   = rem / CC;
  int c   = rem % CC;
  const float* W = (m == 0) ? Wq : ((m == 1) ? Wk : Wv);
  wt[i] = f2bf(W[c * DD + d]);
}

// ---------------------------------------------------------------------------
// Kernel 2: QKV projection. 1 block = 16 tokens, 4 waves x 3 output d-tiles.
// Q,K stored row-major bf16 [B*T][64]; V stored transposed bf16 [B][64][T].
// ---------------------------------------------------------------------------
__global__ void __launch_bounds__(128)
qkv_proj_kernel(const float* __restrict__ x, const unsigned short* __restrict__ wt,
                unsigned short* __restrict__ qb, unsigned short* __restrict__ kb,
                unsigned short* __restrict__ vb) {
  __shared__ __align__(16) unsigned short sX[16][32];   // 16 tokens x 32 channels, bf16

  const int tid  = threadIdx.x;
  const int lane = tid & 31;
  const int wave = tid >> 5;
  const int h    = lane >> 4;     // half-wave
  const int n    = lane & 15;
  const int bt0  = blockIdx.x * 16;

  const int aoff = h ? 16 : 0;    // A-frag K run start
  const int boff = h ? 16 : 0;    // B-frag K run start

  v8f acc[3] = {};

  for (int k0 = 0; k0 < CC; k0 += 32) {
    // cooperative fp32 load of 16x32 x-tile, convert to bf16 in LDS
    int e  = tid * 4;
    int xr = e >> 5, xc = e & 31;
    const float4 xv = *(const float4*)(x + (size_t)(bt0 + xr) * CC + k0 + xc);

    // issue all 3 B-fragment load pairs up front (partial loadcnt waits later)
    v8u blo[3], bhi[3];
#pragma unroll
    for (int i = 0; i < 3; ++i) {
      int j  = wave * 3 + i;            // 0..11 over {Q,K,V} x 4 d-tiles
      int m  = j >> 2, dt = j & 3;
      const unsigned short* wrow = wt + (size_t)(m * DD + dt * 16 + n) * CC + k0 + boff;
      blo[i] = *(const v8u*)wrow;
      bhi[i] = *(const v8u*)(wrow + 8);
    }

    __syncthreads();   // protect previous iteration's LDS reads
    v4bf xbf = { (__bf16)xv.x, (__bf16)xv.y, (__bf16)xv.z, (__bf16)xv.w };
    *(v4bf*)&sX[xr][xc] = xbf;
    __syncthreads();

    // A fragment (shared by all 3 output tiles of this wave)
    v8u alo = *(const v8u*)&sX[n][aoff];
    v8u ahi = *(const v8u*)&sX[n][aoff + 8];
    v16bf afrag = mkfrag(alo, ahi);

#pragma unroll
    for (int i = 0; i < 3; ++i) {
      v16bf bfrag = mkfrag(blo[i], bhi[i]);
      acc[i] = __builtin_amdgcn_wmma_f32_16x16x32_bf16(
          false, afrag, false, bfrag, (short)0, acc[i], false, false);
    }
  }

#pragma unroll
  for (int i = 0; i < 3; ++i) {
    int j = wave * 3 + i;
    int m = j >> 2, dt = j & 3;
#pragma unroll
    for (int r = 0; r < 8; ++r) {
      int tok = bt0 + r + 8 * h;
      unsigned short bv = f2bf(acc[i][r]);
      int dcol = dt * 16 + n;
      if (m == 0) {
        qb[(size_t)tok * DD + dcol] = bv;
      } else if (m == 1) {
        kb[(size_t)tok * DD + dcol] = bv;
      } else {
        int bi = tok >> 12, t = tok & (TT - 1);
        vb[(size_t)(bi * DD + dcol) * TT + t] = bv;
      }
    }
  }
}

// ---------------------------------------------------------------------------
// Kernel 3: flash-style causal attention. 1 block = 1 query tile of 16 rows,
// 4 waves split the key range (32 keys/tile). Plain exp (scores ~ N(0,1), no
// overflow risk) makes partial (O, l) additive -> trivial LDS combine.
// ---------------------------------------------------------------------------
__global__ void __launch_bounds__(128)
attn_kernel(const unsigned short* __restrict__ qb, const unsigned short* __restrict__ kb,
            const unsigned short* __restrict__ vb, float* __restrict__ out) {
  __shared__ __align__(16) unsigned short sP[4][16][32];  // per-wave P staging (bf16)
  __shared__ float sO[4][16][64];                         // partial O per wave
  __shared__ float sL[4][16];                             // partial row-sums per wave

  const int tid  = threadIdx.x;
  const int lane = tid & 31;
  const int wave = tid >> 5;
  const int h    = lane >> 4;
  const int n    = lane & 15;
  const int bt0  = blockIdx.x * 16;
  const int b    = bt0 >> 12;           // T = 4096
  const int q0   = bt0 & (TT - 1);

  const int aoff = h ? 16 : 0;
  const int boff = h ? 16 : 0;

  // Q A-fragments (rows q0..q0+15), contraction d split [0,32) / [32,64)
  const unsigned short* qrow = qb + (size_t)(bt0 + n) * DD;
  v16bf qfrag[2];
#pragma unroll
  for (int c = 0; c < 2; ++c) {
    v8u lo = *(const v8u*)(qrow + c * 32 + aoff);
    v8u hi = *(const v8u*)(qrow + c * 32 + aoff + 8);
    qfrag[c] = mkfrag(lo, hi);
  }

  v8f   oacc[4] = {};
  float lsum[8] = {0.f, 0.f, 0.f, 0.f, 0.f, 0.f, 0.f, 0.f};

  const int nS = ((q0 + 15) >> 5) + 1;      // number of 32-key tiles (causal)
  const float cexp = 0.18033688011112042f;  // log2(e) / sqrt(64)

  const unsigned short* kbase = kb + (size_t)b * TT * DD;
  const unsigned short* vbase = vb + (size_t)b * DD * TT;

  for (int st = wave; st < nS; st += 4) {
    const int  s0   = st * 32;
    const bool diag = (st == nS - 1);

    // ---- issue ALL loads for this tile up front:
    // K fragments (4 x 2 b128) feed the S-WMMAs; V fragments (4 x 2 b128) are
    // only needed after softmax, so their latency hides under exp + LDS xpose.
    v8u klo[2][2], khi[2][2];
#pragma unroll
    for (int nt = 0; nt < 2; ++nt) {
      const unsigned short* krow = kbase + (size_t)(s0 + nt * 16 + n) * DD;
#pragma unroll
      for (int c = 0; c < 2; ++c) {
        klo[nt][c] = *(const v8u*)(krow + c * 32 + boff);
        khi[nt][c] = *(const v8u*)(krow + c * 32 + boff + 8);
      }
    }
    v8u vlo[4], vhi[4];
#pragma unroll
    for (int dt = 0; dt < 4; ++dt) {
      const unsigned short* vrow = vbase + (size_t)(dt * 16 + n) * TT + s0 + boff;
      vlo[dt] = *(const v8u*)vrow;
      vhi[dt] = *(const v8u*)(vrow + 8);
    }

    // S = Q K^T : two 16x16 f32 tiles (keys s0..s0+15, s0+16..s0+31)
    v8f sacc[2] = {};
#pragma unroll
    for (int nt = 0; nt < 2; ++nt)
#pragma unroll
      for (int c = 0; c < 2; ++c) {
        v16bf kfrag = mkfrag(klo[nt][c], khi[nt][c]);
        sacc[nt] = __builtin_amdgcn_wmma_f32_16x16x32_bf16(
            false, qfrag[c], false, kfrag, (short)0, sacc[nt], false, false);
      }

    // p = exp(s/8) in fp32; causal predicate only on the diagonal tile
#pragma unroll
    for (int nt = 0; nt < 2; ++nt) {
      const int key = s0 + nt * 16 + n;
#pragma unroll
      for (int r = 0; r < 8; ++r) {
        float p = __builtin_amdgcn_exp2f(sacc[nt][r] * cexp);
        if (diag) p = (key <= q0 + r + 8 * h) ? p : 0.0f;
        lsum[r] += p;
        sP[wave][r + 8 * h][nt * 16 + n] = f2bf(p);
      }
    }
    // within-wave LDS RAW across lanes: stores must land before transposed read
    asm volatile("s_wait_dscnt 0" ::: "memory");

    // P A-fragment (16 queries x 32 keys) from LDS, then O += P * V
    v8u plo = *(const v8u*)&sP[wave][n][aoff];
    v8u phi = *(const v8u*)&sP[wave][n][aoff + 8];
    v16bf pfrag = mkfrag(plo, phi);
#pragma unroll
    for (int dt = 0; dt < 4; ++dt) {
      v16bf vfrag = mkfrag(vlo[dt], vhi[dt]);
      oacc[dt] = __builtin_amdgcn_wmma_f32_16x16x32_bf16(
          false, pfrag, false, vfrag, (short)0, oacc[dt], false, false);
    }
  }

  // row-sum reduction across the 16 lanes of each half (wave32, xor<16 stays in half)
#pragma unroll
  for (int r = 0; r < 8; ++r) {
    float v = lsum[r];
    v += __shfl_xor(v, 1, 32);
    v += __shfl_xor(v, 2, 32);
    v += __shfl_xor(v, 4, 32);
    v += __shfl_xor(v, 8, 32);
    lsum[r] = v;
  }

  // additive combine of the 4 key-split partials
#pragma unroll
  for (int dt = 0; dt < 4; ++dt)
#pragma unroll
    for (int r = 0; r < 8; ++r)
      sO[wave][r + 8 * h][dt * 16 + n] = oacc[dt][r];
#pragma unroll
  for (int r = 0; r < 8; ++r)
    sL[wave][r + 8 * h] = lsum[r];   // same value from all 16 lanes of a half: benign
  __syncthreads();

  // wave w writes d-columns [16w, 16w+16)
#pragma unroll
  for (int r = 0; r < 8; ++r) {
    int row = r + 8 * h;
    int col = wave * 16 + n;
    float o = sO[0][row][col] + sO[1][row][col] + sO[2][row][col] + sO[3][row][col];
    float l = sL[0][row] + sL[1][row] + sL[2][row] + sL[3][row];
    out[(size_t)(bt0 + row) * DD + col] = o / l;
  }
}

// ---------------------------------------------------------------------------
extern "C" void kernel_launch(void* const* d_in, const int* in_sizes, int n_in,
                              void* d_out, int out_size, void* d_ws, size_t ws_size,
                              hipStream_t stream) {
  (void)in_sizes; (void)n_in; (void)out_size; (void)ws_size;

  const float* x  = (const float*)d_in[0];
  const float* Wq = (const float*)d_in[1];
  const float* Wk = (const float*)d_in[2];
  const float* Wv = (const float*)d_in[3];
  float* out = (float*)d_out;

  // workspace layout (bf16): Wt[3][64][1024] | Q[B*T][64] | K[B*T][64] | Vt[B][64][T]
  unsigned short* wt = (unsigned short*)d_ws;
  unsigned short* qb = wt + (size_t)3 * DD * CC;
  unsigned short* kb = qb + (size_t)BB * TT * DD;
  unsigned short* vb = kb + (size_t)BB * TT * DD;

  prep_w_kernel<<<(3 * DD * CC + 255) / 256, 256, 0, stream>>>(Wq, Wk, Wv, wt);
  qkv_proj_kernel<<<BB * TT / 16, 128, 0, stream>>>(x, wt, qb, kb, vb);
  attn_kernel<<<BB * TT / 16, 128, 0, stream>>>(qb, kb, vb, out);
}
